// GPMatchEncoderTRT_22806276342334
// MI455X (gfx1250) — matrix operand order repaired and checked by reading, not compiled, not run on hardware
//
#include <hip/hip_runtime.h>
#include <hip/hip_bf16.h>

typedef __attribute__((ext_vector_type(16))) _Float16 v16h;
typedef __attribute__((ext_vector_type(8)))  _Float16 v8h;
typedef __attribute__((ext_vector_type(8)))  float    v8f;

#define BATCH 8
#define CH    256
#define HW    4096      // 64*64
#define BETA  10.0f

#define KT        32                 // keys staged per pipeline stage
#define LDS_ROW   264                // halves per key row (256 + 8 pad -> bank spread)
#define STAGE_H   (KT * LDS_ROW)     // halves per stage buffer
#define STAGE_B   (STAGE_H * 2)      // bytes per stage buffer (16896)
#define NSTAGES   (HW / KT)          // 128

// ---------------------------------------------------------------------------
// Stage 1: L2-normalize along C, convert to f16.  src [B,C,HW] -> dst [B,HW,C]
// ---------------------------------------------------------------------------
__global__ __launch_bounds__(256) void normalize_to_f16(
    const float* __restrict__ src, _Float16* __restrict__ dst) {
  int gid = blockIdx.x * blockDim.x + threadIdx.x;   // over B*HW
  int b = gid >> 12;
  int n = gid & (HW - 1);
  const float* s = src + (size_t)b * CH * HW + n;

  float sum = 0.f;
#pragma unroll 8
  for (int c = 0; c < CH; ++c) {
    float v = s[(size_t)c * HW];
    sum = fmaf(v, v, sum);
  }
  float inv = rsqrtf(fmaxf(sum, 1e-12f));            // EPS^2

  _Float16* d = dst + (size_t)gid * CH;
#pragma unroll 4
  for (int c0 = 0; c0 < CH; c0 += 8) {
    v8h h;
#pragma unroll
    for (int i = 0; i < 8; ++i)
      h[i] = (_Float16)(s[(size_t)(c0 + i) * HW] * inv);
    *(v8h*)(d + c0) = h;
  }
}

// ---------------------------------------------------------------------------
// Async stage: copy KT=32 key rows (512B each) global -> LDS, distributed over
// the whole block.  Each thread issues 4 x 16B async copies (ASYNCcnt += 4/wave).
// LDS rows padded to 528B so fragment reads are bank-spread.
// ---------------------------------------------------------------------------
__device__ __forceinline__ void stage_keys_async(const _Float16* __restrict__ Bb,
                                                 int ktile, unsigned lds_dst,
                                                 int t) {
#pragma unroll
  for (int j = 0; j < 4; ++j) {
    int chunk = j * 256 + t;                 // 1024 x 16B chunks, lanes coalesced
    int key = chunk >> 5;                    // 32 chunks per key row
    int c16 = chunk & 31;                    // 16B unit within row
    const void* g = Bb + (size_t)(ktile * KT + key) * CH + c16 * 8;
    unsigned l = lds_dst + (unsigned)(key * (LDS_ROW * 2) + c16 * 16);
    asm volatile("global_load_async_to_lds_b128 %0, %1, off"
                 :: "v"(l), "v"(g) : "memory");
  }
}

// ---------------------------------------------------------------------------
// Stage 2: fused cosine-sim WMMA GEMM + online softmax + coord expectation.
// 8 waves/block, one 16-query stripe per wave; keys double-buffered in LDS.
// ---------------------------------------------------------------------------
__global__ __launch_bounds__(256) void gp_match_wmma(
    const _Float16* __restrict__ Ah,   // [B, HW, C] normalized queries (f16)
    const _Float16* __restrict__ Bh,   // [B, HW, C] normalized keys    (f16)
    float* __restrict__ out) {         // warp [B,2,HW] ++ cert [B,HW]
  __shared__ _Float16 smem[2 * STAGE_H];   // 33,792 B

  const int t    = threadIdx.x;
  const int lane = t & 31;
  const int wv   = t >> 5;
  const int b    = blockIdx.x >> 5;                     // 32 blocks per batch
  const int qbase = ((blockIdx.x & 31) * 8 + wv) * 16;  // 16-query stripe

  const _Float16* Ab = Ah + (size_t)b * HW * CH;
  const _Float16* Bb = Bh + (size_t)b * HW * CH;

  // --- A fragments for full K=256, held in registers (ISA 16x32 f16 layout:
  // lane&15 = row M; low half K {k..k+7, k+16..k+23}, high half K {k+8..,k+24..}).
  v16h afrag[8];
  {
    const int r    = qbase + (lane & 15);
    const int koff = (lane >> 4) * 8;
    const _Float16* pa = Ab + (size_t)r * CH + koff;
#pragma unroll
    for (int ks = 0; ks < 8; ++ks) {
      v8h lo = *(const v8h*)(pa + ks * 32);
      v8h hi = *(const v8h*)(pa + ks * 32 + 16);
      v16h f;
#pragma unroll
      for (int i = 0; i < 8; ++i) { f[i] = lo[i]; f[i + 8] = hi[i]; }
      afrag[ks] = f;
    }
  }

  // Lane-local online-softmax state (row i / i+8, column subset = lane&15).
  float m[8], l[8], ax[8], ay[8];
#pragma unroll
  for (int i = 0; i < 8; ++i) { m[i] = -3.0e38f; l[i] = 0.f; ax[i] = 0.f; ay[i] = 0.f; }

  const int col = lane & 15;          // key column within 16-wide subtile
  const int bko = (lane >> 4) * 8;    // B fragment: low half K {0..7,16..23}, high half {8..15,24..31}

  const unsigned smem_base = (unsigned)(uintptr_t)(&smem[0]);  // LDS byte offset

  stage_keys_async(Bb, 0, smem_base, t);                       // prologue

  for (int kt = 0; kt < NSTAGES; ++kt) {
    if (kt + 1 < NSTAGES) {
      stage_keys_async(Bb, kt + 1, smem_base + ((kt + 1) & 1) * STAGE_B, t);
      asm volatile("s_wait_asynccnt 4" ::: "memory");   // stage kt landed
    } else {
      asm volatile("s_wait_asynccnt 0" ::: "memory");
    }
    __syncthreads();                                    // all waves' copies visible

    const _Float16* sk = smem + (kt & 1) * STAGE_H;

#pragma unroll
    for (int st = 0; st < 2; ++st) {                    // two 16-key subtiles
      const _Float16* pb = sk + (size_t)(st * 16 + col) * LDS_ROW + bko;
      v8f c = {};
#pragma unroll
      for (int ks = 0; ks < 8; ++ks) {
        v8h lo = *(const v8h*)(pb + ks * 32);
        v8h hi = *(const v8h*)(pb + ks * 32 + 16);
        v16h bf;
#pragma unroll
        for (int i = 0; i < 8; ++i) { bf[i] = lo[i]; bf[i + 8] = hi[i]; }
        c = __builtin_amdgcn_wmma_f32_16x16x32_f16(
                false, afrag[ks], false, bf, (short)0, c, false, false);
      }

      const int   n  = kt * KT + st * 16 + col;
      const float xf = (float)(n & 63);
      const float yf = (float)(n >> 6);
#pragma unroll
      for (int i = 0; i < 8; ++i) {
        float s = c[i] * BETA;
        float d = s - m[i];
        float e = __expf(-__builtin_fabsf(d));   // single exp per score
        bool  gt = d > 0.f;
        float p  = gt ? 1.f : e;                 // exp(s - m_new)
        float sc = gt ? e : 1.f;                 // exp(m_old - m_new)
        m[i]  = gt ? s : m[i];
        l[i]  = fmaf(l[i],  sc, p);
        ax[i] = fmaf(ax[i], sc, p * xf);
        ay[i] = fmaf(ay[i], sc, p * yf);
      }
    }
    __syncthreads();                  // buffer (kt&1) reusable two stages later
  }

  // --- merge 16 lane-local softmax states per row (butterfly logsumexp;
  // xor 1,2,4,8 stays inside each 16-lane half -> rows 0-7 / 8-15).
#pragma unroll
  for (int off = 1; off < 16; off <<= 1) {
#pragma unroll
    for (int i = 0; i < 8; ++i) {
      float m2 = __shfl_xor(m[i],  off, 32);
      float l2 = __shfl_xor(l[i],  off, 32);
      float x2 = __shfl_xor(ax[i], off, 32);
      float y2 = __shfl_xor(ay[i], off, 32);
      float mn = fmaxf(m[i], m2);
      float s1 = __expf(m[i] - mn);
      float s2 = __expf(m2   - mn);
      m[i]  = mn;
      l[i]  = l[i]  * s1 + l2 * s2;
      ax[i] = ax[i] * s1 + x2 * s2;
      ay[i] = ay[i] * s1 + y2 * s2;
    }
  }

  if (col == 0) {                               // lanes 0 / 16 write rows 0-7 / 8-15
    const int rbase = qbase + (lane >> 4) * 8;
    float* warp = out;                          // [B, 2, HW]
    float* cert = out + (size_t)BATCH * 2 * HW; // [B, HW]
#pragma unroll
    for (int i = 0; i < 8; ++i) {
      int   q    = rbase + i;
      float invl = 1.0f / l[i];
      warp[((size_t)b * 2    ) * HW + q] = ax[i] * invl;   // expected x
      warp[((size_t)b * 2 + 1) * HW + q] = ay[i] * invl;   // expected y
      cert[(size_t)b * HW + q] = invl;                     // max prob = 1/l
    }
  }
}

// ---------------------------------------------------------------------------
extern "C" void kernel_launch(void* const* d_in, const int* in_sizes, int n_in,
                              void* d_out, int out_size, void* d_ws, size_t ws_size,
                              hipStream_t stream) {
  const float* f0 = (const float*)d_in[0];
  const float* f1 = (const float*)d_in[1];
  float* out = (float*)d_out;

  _Float16* Ah = (_Float16*)d_ws;                       // 16.8 MB
  _Float16* Bh = Ah + (size_t)BATCH * HW * CH;          // 16.8 MB

  normalize_to_f16<<<(BATCH * HW) / 256, 256, 0, stream>>>(f0, Ah);
  normalize_to_f16<<<(BATCH * HW) / 256, 256, 0, stream>>>(f1, Bh);

  // 2048 query stripes, 8 waves (stripes) per 256-thread block
  gp_match_wmma<<<(BATCH * (HW / 16)) / 8, 256, 0, stream>>>(Ah, Bh, out);
}
